// BitwiseTasNet_77644418777121
// MI455X (gfx1250) — matrix operand order repaired and checked by reading, not compiled, not run on hardware
//
#include <hip/hip_runtime.h>
#include <hip/hip_bf16.h>

typedef __attribute__((ext_vector_type(16))) _Float16 v16h;
typedef __attribute__((ext_vector_type(8)))  _Float16 v8h;
typedef __attribute__((ext_vector_type(8)))  float    v8f;

// ---- model dims (match reference) ----
constexpr int N_  = 4;
constexpr int T_  = 8000;
constexpr int C_  = 256;   // bottleneck channels
constexpr int D_  = 512;   // dconv channels
constexpr int KK  = 3;     // depthwise kernel
constexpr int FK  = 20;    // front kernel
constexpr int FS  = 10;    // front stride
constexpr int BLOCKS = 8;
constexpr int NB  = 32;    // 4 repeats * 8 blocks
constexpr float EPS = 1e-5f;

// encoder output length: pad FK both sides, k=FK, stride=FS
constexpr int L_ = (T_ + 2 * FK - (FK - 1) - 1) / FS + 1;   // 803
constexpr int LT = (L_ + 15) / 16;                          // 51 column tiles per n

// padded LDS column strides (halves) to dodge bank conflicts; keep 32B align
constexpr int BS1 = C_ + 16;   // 272 (272*2 = 544 B, multiple of 32)
constexpr int BS2 = D_ + 16;   // 528 (1056 B, multiple of 32)

// ---------------- prep kernels ----------------

__global__ void convert_weights_kernel(const float* __restrict__ w1,
                                       const float* __restrict__ w2,
                                       _Float16* __restrict__ w1h,
                                       _Float16* __restrict__ w2h) {
    const int M = NB * D_ * C_;
    int idx = blockIdx.x * blockDim.x + threadIdx.x;
    if (idx < M)            w1h[idx]     = (_Float16)w1[idx];
    else if (idx < 2 * M)   w2h[idx - M] = (_Float16)w2[idx - M];
}

__global__ void bnprep_kernel(const float* __restrict__ g1, const float* __restrict__ b1,
                              const float* __restrict__ m1, const float* __restrict__ v1,
                              const float* __restrict__ g2, const float* __restrict__ b2,
                              const float* __restrict__ m2, const float* __restrict__ v2,
                              const float* __restrict__ g3, const float* __restrict__ b3,
                              const float* __restrict__ m3, const float* __restrict__ v3,
                              float* __restrict__ s1, float* __restrict__ t1,
                              float* __restrict__ s2, float* __restrict__ t2,
                              float* __restrict__ s3, float* __restrict__ t3) {
    const int n1 = NB * C_, n2 = NB * D_;
    int idx = blockIdx.x * blockDim.x + threadIdx.x;
    if (idx < n1) {
        float s = g1[idx] * rsqrtf(v1[idx] + EPS);
        s1[idx] = s; t1[idx] = b1[idx] - m1[idx] * s;
    } else if (idx < n1 + n2) {
        int j = idx - n1;
        float s = g2[j] * rsqrtf(v2[j] + EPS);
        s2[j] = s; t2[j] = b2[j] - m2[j] * s;
    } else if (idx < n1 + 2 * n2) {
        int j = idx - n1 - n2;
        float s = g3[j] * rsqrtf(v3[j] + EPS);
        s3[j] = s; t3[j] = b3[j] - m3[j] * s;
    }
}

// ---------------- encoder ----------------
// xe[n,c,l] = sum_k w_enc[c,0,k] * x[n,0, l*FS + k - FK]
__global__ void encoder_kernel(const float* __restrict__ x,
                               const float* __restrict__ w_enc,
                               float* __restrict__ xe, float* __restrict__ h) {
    int idx = blockIdx.x * blockDim.x + threadIdx.x;
    if (idx >= N_ * C_ * L_) return;
    int l = idx % L_;
    int c = (idx / L_) % C_;
    int n = idx / (L_ * C_);
    const float* xp = x + (size_t)n * T_;
    const float* wp = w_enc + c * FK;
    float acc = 0.f;
    int base = l * FS - FK;
#pragma unroll
    for (int k = 0; k < FK; ++k) {
        int p = base + k;
        if (p >= 0 && p < T_) acc += wp[k] * xp[p];
    }
    xe[idx] = acc;
    h[idx]  = acc;
}

// ---------------- GEMM1: r2 = bn2( W1 @ bn1(h) ) ----------------
// h:[N,C,L] f32, w1h:[D,C] f16 (this block), r2:[N,D,L] f32
__global__ __launch_bounds__(256)
void gemm1_kernel(const float* __restrict__ h, const _Float16* __restrict__ w1h,
                  const float* __restrict__ s1, const float* __restrict__ t1,
                  const float* __restrict__ s2, const float* __restrict__ t2,
                  float* __restrict__ r2) {
    __shared__ _Float16 Bs[16 * BS1];   // col-major activation tile (f16, bn1 applied)
    const int tileId = blockIdx.x;
    const int n  = tileId / LT;
    const int l0 = (tileId % LT) * 16;
    const int tid = threadIdx.x;

    // stage B tile: 16 cols x C K-values
    {
        const int col = tid & 15;
        const int kc  = tid >> 4;           // 0..15, 16 K each
        const int l   = l0 + col;
        const bool valid = (l < L_);
        const float* hp = h + (size_t)n * C_ * L_ + l;
#pragma unroll
        for (int kk = 0; kk < 16; ++kk) {
            const int k = kc * 16 + kk;
            float v = valid ? (hp[(size_t)k * L_] * s1[k] + t1[k]) : 0.0f;
            Bs[col * BS1 + k] = (_Float16)v;
        }
    }
    __syncthreads();

    const int wave = tid >> 5;
    const int lane = tid & 31;
    const int rowA = lane & 15;
    const int kbA  = (lane < 16) ? 0 : 16;
    const int kbB  = (lane < 16) ? 0 : 16;
    const int colq = lane & 15;
    const int rsel = (lane >> 4) << 3;      // +0 or +8 output rows

#pragma unroll
    for (int rep = 0; rep < 4; ++rep) {
        const int dt = wave + rep * 8;      // d-tile 0..31
        const _Float16* wrow = w1h + (size_t)(dt * 16 + rowA) * C_;
        v8f acc = {};
#pragma unroll
        for (int k0 = 0; k0 < C_; k0 += 32) {
            v8h alo = *(const v8h*)(wrow + k0 + kbA);
            v8h ahi = *(const v8h*)(wrow + k0 + 8 + kbA);
            v16h a;
#pragma unroll
            for (int t = 0; t < 8; ++t) { a[t] = alo[t]; a[8 + t] = ahi[t]; }
            v16h b = *(const v16h*)(&Bs[colq * BS1 + k0 + kbB]);
            acc = __builtin_amdgcn_wmma_f32_16x16x32_f16(
                      false, a, false, b, (short)0, acc, false, false);
        }
        const int l = l0 + colq;
        if (l < L_) {
            float* rp = r2 + (size_t)n * D_ * L_ + l;
#pragma unroll
            for (int j = 0; j < 8; ++j) {
                const int d = dt * 16 + rsel + j;
                rp[(size_t)d * L_] = acc[j] * s2[d] + t2[d];
            }
        }
    }
}

// ---------------- GEMM2: h += W2 @ bn3(depthwise_dil(r2)) ----------------
// r2:[N,D,L] f32, w2h:[C,D] f16 (this block), wd:[D,3] f32, h updated in place
__global__ __launch_bounds__(256)
void gemm2_kernel(const float* __restrict__ r2, const _Float16* __restrict__ w2h,
                  const float* __restrict__ wd,
                  const float* __restrict__ s3, const float* __restrict__ t3,
                  float* __restrict__ h, int dil) {
    __shared__ _Float16 Bs[16 * BS2];
    const int tileId = blockIdx.x;
    const int n  = tileId / LT;
    const int l0 = (tileId % LT) * 16;
    const int tid = threadIdx.x;

    // stage B tile: fuse 3-tap dilated depthwise conv + bn3, f32 -> f16
    {
        const int col = tid & 15;
        const int l   = l0 + col;
        const bool valid = (l < L_);
        const float* rp = r2 + (size_t)n * D_ * L_;
        const int dbase = (tid >> 4) * 32;          // 32 d per thread
#pragma unroll
        for (int dd = 0; dd < 32; ++dd) {
            const int d = dbase + dd;
            float v = 0.f;
            if (valid) {
                const float* rd = rp + (size_t)d * L_;
                const float w0 = wd[d * KK + 0];
                const float w1 = wd[d * KK + 1];
                const float w2 = wd[d * KK + 2];
                float a = w1 * rd[l];
                const int lm = l - dil, lp = l + dil;
                if (lm >= 0) a += w0 * rd[lm];
                if (lp < L_) a += w2 * rd[lp];
                v = a * s3[d] + t3[d];
            }
            Bs[col * BS2 + d] = (_Float16)v;
        }
    }
    __syncthreads();

    const int wave = tid >> 5;
    const int lane = tid & 31;
    const int rowA = lane & 15;
    const int kbA  = (lane < 16) ? 0 : 16;
    const int kbB  = (lane < 16) ? 0 : 16;
    const int colq = lane & 15;
    const int rsel = (lane >> 4) << 3;

#pragma unroll
    for (int rep = 0; rep < 2; ++rep) {
        const int ct = wave + rep * 8;              // c-tile 0..15
        const _Float16* wrow = w2h + (size_t)(ct * 16 + rowA) * D_;
        v8f acc = {};
#pragma unroll
        for (int k0 = 0; k0 < D_; k0 += 32) {
            v8h alo = *(const v8h*)(wrow + k0 + kbA);
            v8h ahi = *(const v8h*)(wrow + k0 + 8 + kbA);
            v16h a;
#pragma unroll
            for (int t = 0; t < 8; ++t) { a[t] = alo[t]; a[8 + t] = ahi[t]; }
            v16h b = *(const v16h*)(&Bs[colq * BS2 + k0 + kbB]);
            acc = __builtin_amdgcn_wmma_f32_16x16x32_f16(
                      false, a, false, b, (short)0, acc, false, false);
        }
        const int l = l0 + colq;
        if (l < L_) {
            float* hp = h + (size_t)n * C_ * L_ + l;
#pragma unroll
            for (int j = 0; j < 8; ++j) {
                const int c = ct * 16 + rsel + j;
                hp[(size_t)c * L_] += acc[j];       // residual add in place
            }
        }
    }
}

// ---------------- mask: y = xe * sigmoid(h) ----------------
__global__ void mask_kernel(const float* __restrict__ xe, const float* __restrict__ h,
                            float* __restrict__ y) {
    int idx = blockIdx.x * blockDim.x + threadIdx.x;
    if (idx >= N_ * C_ * L_) return;
    float hv = h[idx];
    y[idx] = xe[idx] * (1.0f / (1.0f + __expf(-hv)));
}

// ---------------- decoder (transpose conv, 2 taps per output sample) ----------------
// out[n,t] = sum_c w_dec[c,0,j0]*y[n,c,t/FS+2] + w_dec[c,0,j0+FS]*y[n,c,t/FS+1], j0=t%FS
__global__ void decoder_kernel(const float* __restrict__ y, const float* __restrict__ w_dec,
                               float* __restrict__ out) {
    int idx = blockIdx.x * blockDim.x + threadIdx.x;
    if (idx >= N_ * T_) return;
    int t = idx % T_;
    int n = idx / T_;
    int j0 = t % FS;
    int l1 = t / FS + 2;   // always < L_
    int l2 = t / FS + 1;
    const float* yn = y + (size_t)n * C_ * L_;
    float acc = 0.f;
    for (int c = 0; c < C_; ++c) {
        const float* wc = w_dec + c * FK;
        const float* yc = yn + (size_t)c * L_;
        acc += wc[j0] * yc[l1] + wc[j0 + FS] * yc[l2];
    }
    out[idx] = acc;
}

// ---------------- host launch ----------------

static constexpr size_t alignUp(size_t x) { return (x + 255) & ~size_t(255); }

extern "C" void kernel_launch(void* const* d_in, const int* in_sizes, int n_in,
                              void* d_out, int out_size, void* d_ws, size_t ws_size,
                              hipStream_t stream) {
    const float* x     = (const float*)d_in[0];
    const float* w_enc = (const float*)d_in[1];
    const float* w1    = (const float*)d_in[2];
    const float* wd    = (const float*)d_in[3];
    const float* w2    = (const float*)d_in[4];
    const float* w_dec = (const float*)d_in[5];
    const float* bn1g  = (const float*)d_in[6];
    const float* bn1b  = (const float*)d_in[7];
    const float* bn1m  = (const float*)d_in[8];
    const float* bn1v  = (const float*)d_in[9];
    const float* bn2g  = (const float*)d_in[10];
    const float* bn2b  = (const float*)d_in[11];
    const float* bn2m  = (const float*)d_in[12];
    const float* bn2v  = (const float*)d_in[13];
    const float* bn3g  = (const float*)d_in[14];
    const float* bn3b  = (const float*)d_in[15];
    const float* bn3m  = (const float*)d_in[16];
    const float* bn3v  = (const float*)d_in[17];
    float* out = (float*)d_out;

    // workspace carve-up
    char* ws = (char*)d_ws;
    size_t off = 0;
    constexpr size_t WSZ = (size_t)NB * D_ * C_;             // elements per f16 weight set
    _Float16* w1h = (_Float16*)(ws + off); off += alignUp(WSZ * sizeof(_Float16));
    _Float16* w2h = (_Float16*)(ws + off); off += alignUp(WSZ * sizeof(_Float16));
    float* s1 = (float*)(ws + off); off += alignUp((size_t)NB * C_ * sizeof(float));
    float* t1 = (float*)(ws + off); off += alignUp((size_t)NB * C_ * sizeof(float));
    float* s2 = (float*)(ws + off); off += alignUp((size_t)NB * D_ * sizeof(float));
    float* t2 = (float*)(ws + off); off += alignUp((size_t)NB * D_ * sizeof(float));
    float* s3 = (float*)(ws + off); off += alignUp((size_t)NB * D_ * sizeof(float));
    float* t3 = (float*)(ws + off); off += alignUp((size_t)NB * D_ * sizeof(float));
    float* xe = (float*)(ws + off); off += alignUp((size_t)N_ * C_ * L_ * sizeof(float));
    float* h  = (float*)(ws + off); off += alignUp((size_t)N_ * C_ * L_ * sizeof(float));
    float* r2 = (float*)(ws + off); off += alignUp((size_t)N_ * D_ * L_ * sizeof(float));
    float* y  = (float*)(ws + off); off += alignUp((size_t)N_ * C_ * L_ * sizeof(float));
    (void)ws_size; (void)n_in; (void)in_sizes; (void)out_size;

    // 1) weight conversion + bn folding
    {
        int total = 2 * (int)WSZ;
        convert_weights_kernel<<<(total + 255) / 256, 256, 0, stream>>>(w1, w2, w1h, w2h);
        int bntot = NB * (C_ + 2 * D_);
        bnprep_kernel<<<(bntot + 255) / 256, 256, 0, stream>>>(
            bn1g, bn1b, bn1m, bn1v, bn2g, bn2b, bn2m, bn2v, bn3g, bn3b, bn3m, bn3v,
            s1, t1, s2, t2, s3, t3);
    }

    // 2) encoder
    {
        int total = N_ * C_ * L_;
        encoder_kernel<<<(total + 255) / 256, 256, 0, stream>>>(x, w_enc, xe, h);
    }

    // 3) 32 residual blocks, two WMMA GEMM kernels each
    const int tiles = N_ * LT;    // 204 column tiles
    for (int i = 0; i < NB; ++i) {
        const int dil = 1 << (i % BLOCKS);
        gemm1_kernel<<<tiles, 256, 0, stream>>>(
            h, w1h + (size_t)i * D_ * C_, s1 + (size_t)i * C_, t1 + (size_t)i * C_,
            s2 + (size_t)i * D_, t2 + (size_t)i * D_, r2);
        gemm2_kernel<<<tiles, 256, 0, stream>>>(
            r2, w2h + (size_t)i * C_ * D_, wd + (size_t)i * D_ * KK,
            s3 + (size_t)i * D_, t3 + (size_t)i * D_, h, dil);
    }

    // 4) mask + decoder
    {
        int total = N_ * C_ * L_;
        mask_kernel<<<(total + 255) / 256, 256, 0, stream>>>(xe, h, y);
        int dt = N_ * T_;
        decoder_kernel<<<(dt + 255) / 256, 256, 0, stream>>>(y, w_dec, out);
    }
}
